// QuantizedLinear_12214886990081
// MI455X (gfx1250) — compile-verified, matrix operand order checked by reading, not run on hardware
//
#include <hip/hip_runtime.h>

typedef __attribute__((ext_vector_type(16))) __bf16 v16bf;
typedef __attribute__((ext_vector_type(8)))  __bf16 v8bf;
typedef __attribute__((ext_vector_type(4)))  __bf16 v4bf;
typedef __attribute__((ext_vector_type(8)))  float  v8f;

#define IN_F   4096
#define OUT_F  4096
#define BM     128
#define BN     128
#define KT     32
#define LDSS   40          // 32 + 8 pad elements -> 80B row stride (16B aligned)
#define NKT    (IN_F / KT) // 128

__constant__ float NF4_LEVELS[16] = {
    -1.0f, -0.6961928009986877f, -0.5250730514526367f, -0.39491748809814453f,
    -0.28444138169288635f, -0.18477343022823334f, -0.09105003625154495f, 0.0f,
    0.07958029955625534f, 0.16093020141124725f, 0.24611230194568634f,
    0.33791524171829224f, 0.44070982933044434f, 0.5626170039176941f,
    0.7229568362236023f, 1.0f};

__global__ __launch_bounds__(256, 2)
void nf4_linear_wmma(const float* __restrict__ x,
                     const int*   __restrict__ packed,
                     const float* __restrict__ absmax,
                     const float* __restrict__ bias,
                     float*       __restrict__ out)
{
    __shared__ __bf16 As[BM * LDSS];       // activations, bf16
    __shared__ __bf16 Bs[BN * LDSS];       // dequantized weights, bf16 (LUT only; absmax applied in epilogue)
    __shared__ unsigned int tbl[256];      // byte -> packed pair of bf16 (lo half = high nibble = element 2j)

    const int tid  = threadIdx.x;
    const int lane = tid & 31;
    const int wave = tid >> 5;     // 0..7
    const int hf   = lane >> 4;    // half of wave (0/1)
    const int lr   = lane & 15;

    // Build 256-entry nibble-pair dequant table once per block.
    if (tid < 256) {
        const int b = tid;
        const __bf16 hv = (__bf16)NF4_LEVELS[(b >> 4) & 15];  // element 2j
        const __bf16 lv = (__bf16)NF4_LEVELS[b & 15];         // element 2j+1
        const unsigned int lo = (unsigned int)__builtin_bit_cast(unsigned short, hv);
        const unsigned int hi = (unsigned int)__builtin_bit_cast(unsigned short, lv);
        tbl[b] = lo | (hi << 16);
    }

    const int bn = blockIdx.x * BN;        // output-feature tile base
    const int bm = blockIdx.y * BM;        // row (batch*seq) tile base

    // 8 waves -> 2 (M) x 4 (N); each wave: 64x32 = 4x2 WMMA tiles of 16x16
    const int wm = (wave >> 2) * 64;
    const int wn = (wave & 3) * 32;

    const v8f vzero = {};
    v8f acc[4][2];
#pragma unroll
    for (int i = 0; i < 4; ++i)
#pragma unroll
        for (int j = 0; j < 2; ++j) acc[i][j] = vzero;

    for (int kt = 0; kt < NKT; ++kt) {
        __syncthreads();   // previous compute (or tbl init) done before restaging

        // ---- Stage A: 128x32 fp32 -> bf16 (4 x float4 per thread) ----
#pragma unroll
        for (int it = 0; it < 4; ++it) {
            const int slot = tid + it * 256;          // 0..1023
            const int row  = slot >> 3;               // 0..127
            const int kq   = slot & 7;                // float4 group within K-tile
            const float4 v = *(const float4*)(x + (size_t)(bm + row) * IN_F
                                                + (size_t)kt * KT + kq * 4);
            v4bf bv;
            bv.x = (__bf16)v.x; bv.y = (__bf16)v.y;
            bv.z = (__bf16)v.z; bv.w = (__bf16)v.w;
            *(v4bf*)&As[row * LDSS + kq * 4] = bv;    // ds_store_b64
        }

        // ---- Stage B: dequantize 128x32 NF4 weights via pair LUT (2 x int4 per thread) ----
#pragma unroll
        for (int it = 0; it < 2; ++it) {
            const int slot = tid + it * 256;          // 0..511
            const int row  = slot >> 2;               // 0..127 (output-feature row)
            const int grp  = slot & 3;                // int4 group (8 weights)
            const int4 pv  = *(const int4*)(packed + (size_t)(bn + row) * (IN_F / 2)
                                                   + (size_t)kt * (KT / 2) + grp * 4);
            uint4 w;
            w.x = tbl[pv.x & 255];
            w.y = tbl[pv.y & 255];
            w.z = tbl[pv.z & 255];
            w.w = tbl[pv.w & 255];
            *(uint4*)&Bs[row * LDSS + grp * 8] = w;   // ds_store_b128
        }

        __syncthreads();

        // ---- Compute: load fragments per ISA 16-bit layout, 8 WMMAs ----
        // lane half 0 holds K 0..7 and 16..23; half 1 holds K 8..15 and 24..31
        v16bf afrag[4];
#pragma unroll
        for (int i = 0; i < 4; ++i) {
            const __bf16* p = &As[(wm + i * 16 + lr) * LDSS + hf * 8];
            const v8bf lo = *(const v8bf*)p;           // ds_load_b128
            const v8bf hi = *(const v8bf*)(p + 16);    // ds_load_b128
            afrag[i] = __builtin_shufflevector(lo, hi, 0,1,2,3,4,5,6,7,
                                                       8,9,10,11,12,13,14,15);
        }
        v16bf bfrag[2];
#pragma unroll
        for (int j = 0; j < 2; ++j) {
            const __bf16* p = &Bs[(wn + j * 16 + lr) * LDSS + hf * 8];
            const v8bf lo = *(const v8bf*)p;
            const v8bf hi = *(const v8bf*)(p + 16);
            bfrag[j] = __builtin_shufflevector(lo, hi, 0,1,2,3,4,5,6,7,
                                                       8,9,10,11,12,13,14,15);
        }
#pragma unroll
        for (int i = 0; i < 4; ++i)
#pragma unroll
            for (int j = 0; j < 2; ++j)
                acc[i][j] = __builtin_amdgcn_wmma_f32_16x16x32_bf16(
                    /*neg_a=*/false, afrag[i],
                    /*neg_b=*/false, bfrag[j],
                    /*c_mod=*/(short)0, acc[i][j],
                    /*reuse_a=*/false, /*reuse_b=*/false);
    }

    // ---- Epilogue: out[m][n] = acc * absmax[n] + bias[n] ----
#pragma unroll
    for (int j = 0; j < 2; ++j) {
        const int col = bn + wn + j * 16 + lr;        // output feature (N)
        const float sc = absmax[col];
        const float bi = bias[col];
#pragma unroll
        for (int i = 0; i < 4; ++i) {
            const int m0 = bm + wm + i * 16 + hf * 8; // rows per C/D layout
#pragma unroll
            for (int v = 0; v < 8; ++v) {
                out[(size_t)(m0 + v) * OUT_F + col] = acc[i][j][v] * sc + bi;
            }
        }
    }
}

extern "C" void kernel_launch(void* const* d_in, const int* in_sizes, int n_in,
                              void* d_out, int out_size, void* d_ws, size_t ws_size,
                              hipStream_t stream) {
    const float* x      = (const float*)d_in[0];
    const int*   packed = (const int*)d_in[1];
    const float* absmax = (const float*)d_in[2];
    const float* bias   = (const float*)d_in[3];
    float*       out    = (float*)d_out;

    const int M = in_sizes[0] / IN_F;                 // 4*2048 = 8192 rows
    dim3 grid(OUT_F / BN, M / BM);                    // (32, 64)
    nf4_linear_wmma<<<grid, dim3(256, 1, 1), 0, stream>>>(x, packed, absmax, bias, out);
}
